// TransformerEncoderBlock_24008867185073
// MI455X (gfx1250) — compile-verified, matrix operand order checked by reading, not run on hardware
//
#include <hip/hip_runtime.h>
#include <math.h>

// ---------------------------------------------------------------------------
// TransformerEncoderBlock for MI455X (gfx1250, wave32, WMMA)
//   D=768, S=2048, B=8  -> M = 16384 token rows
//   All GEMMs run on v_wmma_f32_16x16x32_f16 (f16 inputs, f32 accumulate).
// ---------------------------------------------------------------------------

typedef __attribute__((ext_vector_type(16))) _Float16     v16h;
typedef __attribute__((ext_vector_type(8)))  float        v8f;
typedef __attribute__((ext_vector_type(4)))  unsigned int u32x4;
typedef __attribute__((ext_vector_type(4)))  float        f32x4;

#define DMODEL 768
#define SEQ    2048
#define NBATCH 8
#define BM 128
#define BN 128
#define BK 32
#define LDSW 40   // padded LDS row stride in halfs (80B -> 20-word bank stride)

__device__ __forceinline__ float gelu_exact(float x) {
    return 0.5f * x * (1.0f + erff(x * 0.70710678118654752f));
}

// ---------------------------------------------------------------------------
// Tiled WMMA GEMM:  C[M,N] = act( A16[M,K] * B[N,K]^T + bias ) (+ residual)
//   A is always f16 row-major, K contiguous.
//   B is [N,K] row-major; BF32 => f32 in memory, converted to f16 during the
//   global->LDS stage (weights live in L2, conversion is free vs math).
//   OMODE: 0 = f32 out, 1 = f16 out, 2 = f16 out transposed per-batch [D][S]
// ---------------------------------------------------------------------------
template<bool BF32, bool BIAS, bool GELU, bool RES, int OMODE>
__global__ __launch_bounds__(256)
void gemm_wmma_kernel(const _Float16* __restrict__ A,
                      const void*     __restrict__ Bv,
                      const float*    __restrict__ bias,
                      const float*    __restrict__ res,
                      float*          __restrict__ out32,
                      _Float16*       __restrict__ out16,
                      int M, int N, int K,
                      long sA, long sB, long sO, long sRes)
{
    const int z = blockIdx.z;
    A += (size_t)z * sA;
    const _Float16* Bh = nullptr;
    const float*    Bf = nullptr;
    if constexpr (BF32) Bf = (const float*)Bv + (size_t)z * sB;
    else                Bh = (const _Float16*)Bv + (size_t)z * sB;
    if constexpr (RES)        res   += (size_t)z * sRes;
    if constexpr (OMODE == 0) out32 += (size_t)z * sO;
    if constexpr (OMODE == 1) out16 += (size_t)z * sO;

    const int tid    = threadIdx.x;
    const int lane   = tid & 31;
    const int waveId = tid >> 5;          // 8 waves
    const int waveM  = waveId >> 2;       // 0..1  -> 64 rows each
    const int waveN  = waveId & 3;        // 0..3  -> 32 cols each
    const int lr     = lane & 15;
    const int hi     = lane >> 4;         // half-wave select
    const int k0     = hi * 8;            // ISA 16-bit A/B fragment K base

    const int rowM0 = blockIdx.y * BM;
    const int colN0 = blockIdx.x * BN;

    __shared__ alignas(16) _Float16 ldsA[BM * LDSW];
    __shared__ alignas(16) _Float16 ldsB[BN * LDSW];

    v8f acc[4][2];
#pragma unroll
    for (int i = 0; i < 4; ++i)
#pragma unroll
        for (int j = 0; j < 2; ++j) acc[i][j] = (v8f){0.f,0.f,0.f,0.f,0.f,0.f,0.f,0.f};

    // register staging for global->LDS pipelining
    u32x4 ra[2];
    u32x4 rbh[2];
    f32x4 rbf[4];

    auto loadA = [&](int kk) {
#pragma unroll
        for (int jj = 0; jj < 2; ++jj) {
            int idx = tid + jj * 256;
            int r = idx >> 2, c = idx & 3;            // 128 rows x 4 chunks of 8 halfs
            ra[jj] = *(const u32x4*)(A + (size_t)(rowM0 + r) * K + kk + c * 8);
        }
    };
    auto loadB = [&](int kk) {
#pragma unroll
        for (int jj = 0; jj < 2; ++jj) {
            int idx = tid + jj * 256;
            int r = idx >> 2, c = idx & 3;
            if constexpr (BF32) {
                const float* p = Bf + (size_t)(colN0 + r) * K + kk + c * 8;
                rbf[jj * 2 + 0] = *(const f32x4*)p;
                rbf[jj * 2 + 1] = *(const f32x4*)(p + 4);
            } else {
                rbh[jj] = *(const u32x4*)(Bh + (size_t)(colN0 + r) * K + kk + c * 8);
            }
        }
    };
    auto stage = [&]() {
#pragma unroll
        for (int jj = 0; jj < 2; ++jj) {
            int idx = tid + jj * 256;
            int r = idx >> 2, c = idx & 3;
            *(u32x4*)&ldsA[r * LDSW + c * 8] = ra[jj];
            if constexpr (BF32) {
                union { _Float16 hh[8]; u32x4 uu; } pk;
#pragma unroll
                for (int t = 0; t < 4; ++t) {
                    pk.hh[t]     = (_Float16)rbf[jj * 2 + 0][t];
                    pk.hh[4 + t] = (_Float16)rbf[jj * 2 + 1][t];
                }
                *(u32x4*)&ldsB[r * LDSW + c * 8] = pk.uu;
            } else {
                *(u32x4*)&ldsB[r * LDSW + c * 8] = rbh[jj];
            }
        }
    };

    const int nK = K / BK;
    loadA(0);
    loadB(0);

    for (int kt = 0; kt < nK; ++kt) {
        stage();
        __syncthreads();
        if (kt + 1 < nK) { loadA((kt + 1) * BK); loadB((kt + 1) * BK); }

        union { u32x4 u[2]; v16h h; } af[4], bf[2];
#pragma unroll
        for (int i = 0; i < 4; ++i) {
            int r = waveM * 64 + i * 16 + lr;
            af[i].u[0] = *(const u32x4*)&ldsA[r * LDSW + k0];
            af[i].u[1] = *(const u32x4*)&ldsA[r * LDSW + k0 + 16];
        }
#pragma unroll
        for (int j = 0; j < 2; ++j) {
            int r = waveN * 32 + j * 16 + lr;
            bf[j].u[0] = *(const u32x4*)&ldsB[r * LDSW + k0];
            bf[j].u[1] = *(const u32x4*)&ldsB[r * LDSW + k0 + 16];
        }
#pragma unroll
        for (int i = 0; i < 4; ++i)
#pragma unroll
            for (int j = 0; j < 2; ++j)
                acc[i][j] = __builtin_amdgcn_wmma_f32_16x16x32_f16(
                    false, af[i].h, false, bf[j].h, (short)0, acc[i][j], false, false);
        __syncthreads();
    }

    // Epilogue. C/D layout: VGPR v -> M = v + 8*hi, N = lane&15.
#pragma unroll
    for (int i = 0; i < 4; ++i) {
#pragma unroll
        for (int j = 0; j < 2; ++j) {
#pragma unroll
            for (int v = 0; v < 8; ++v) {
                int row = rowM0 + waveM * 64 + i * 16 + hi * 8 + v;
                int col = colN0 + waveN * 32 + j * 16 + lr;
                float val = acc[i][j][v];
                if constexpr (BIAS) val += bias[col];
                if constexpr (GELU) val = gelu_exact(val);
                if constexpr (RES)  val += res[(size_t)row * N + col];
                if constexpr (OMODE == 0) {
                    out32[(size_t)row * N + col] = val;
                } else if constexpr (OMODE == 1) {
                    out16[(size_t)row * N + col] = (_Float16)val;
                } else {
                    // transposed f16 store: V^T laid out as [BATCH][D][S]
                    int b = row >> 11;       // /2048
                    int t = row & 2047;
                    out16[((size_t)b * DMODEL + col) * SEQ + t] = (_Float16)val;
                }
            }
        }
    }
}

// ---------------------------------------------------------------------------
// LayerNorm over D=768, one 256-thread block per row, f16 output
// ---------------------------------------------------------------------------
__global__ __launch_bounds__(256)
void ln_kernel(const float* __restrict__ x, const float* __restrict__ g,
               const float* __restrict__ be, _Float16* __restrict__ out)
{
    const int tid = threadIdx.x;
    const float* xr = x + (size_t)blockIdx.x * DMODEL;
    float v0 = xr[tid], v1 = xr[tid + 256], v2 = xr[tid + 512];

    __shared__ float red[256];
    red[tid] = v0 + v1 + v2;
    __syncthreads();
    for (int o = 128; o > 0; o >>= 1) {
        if (tid < o) red[tid] += red[tid + o];
        __syncthreads();
    }
    float mu = red[0] * (1.0f / DMODEL);
    __syncthreads();

    float d0 = v0 - mu, d1 = v1 - mu, d2 = v2 - mu;
    red[tid] = d0 * d0 + d1 * d1 + d2 * d2;
    __syncthreads();
    for (int o = 128; o > 0; o >>= 1) {
        if (tid < o) red[tid] += red[tid + o];
        __syncthreads();
    }
    float rstd = rsqrtf(red[0] * (1.0f / DMODEL) + 1e-5f);

    _Float16* orow = out + (size_t)blockIdx.x * DMODEL;
    orow[tid]       = (_Float16)(d0 * rstd * g[tid]       + be[tid]);
    orow[tid + 256] = (_Float16)(d1 * rstd * g[tid + 256] + be[tid + 256]);
    orow[tid + 512] = (_Float16)(d2 * rstd * g[tid + 512] + be[tid + 512]);
}

// ---------------------------------------------------------------------------
// Row softmax over S=2048 with 1/sqrt(D) scale folded in; f16 probabilities
// ---------------------------------------------------------------------------
__global__ __launch_bounds__(256)
void softmax_kernel(const float* __restrict__ scores, _Float16* __restrict__ attn)
{
    const int tid = threadIdx.x;
    const float* sr = scores + (size_t)blockIdx.x * SEQ;
    _Float16*    ar = attn   + (size_t)blockIdx.x * SEQ;
    const float scale = 0.03608439182435161f;  // 1/sqrt(768)

    float l[8];
    float m = -3.4e38f;
#pragma unroll
    for (int i = 0; i < 8; ++i) {
        l[i] = sr[tid + i * 256] * scale;
        m = fmaxf(m, l[i]);
    }
    __shared__ float red[256];
    red[tid] = m;
    __syncthreads();
    for (int o = 128; o > 0; o >>= 1) {
        if (tid < o) red[tid] = fmaxf(red[tid], red[tid + o]);
        __syncthreads();
    }
    m = red[0];
    __syncthreads();

    float s = 0.f;
#pragma unroll
    for (int i = 0; i < 8; ++i) {
        l[i] = expf(l[i] - m);
        s += l[i];
    }
    red[tid] = s;
    __syncthreads();
    for (int o = 128; o > 0; o >>= 1) {
        if (tid < o) red[tid] += red[tid + o];
        __syncthreads();
    }
    float inv = 1.0f / red[0];
#pragma unroll
    for (int i = 0; i < 8; ++i) ar[tid + i * 256] = (_Float16)(l[i] * inv);
}

// ---------------------------------------------------------------------------
extern "C" void kernel_launch(void* const* d_in, const int* in_sizes, int n_in,
                              void* d_out, int out_size, void* d_ws, size_t ws_size,
                              hipStream_t stream)
{
    const float* x   = (const float*)d_in[0];
    const float* Wq  = (const float*)d_in[1];
    const float* bq  = (const float*)d_in[2];
    const float* Wk  = (const float*)d_in[3];
    const float* bk  = (const float*)d_in[4];
    const float* Wv  = (const float*)d_in[5];
    const float* bv  = (const float*)d_in[6];
    const float* W1  = (const float*)d_in[7];
    const float* b1  = (const float*)d_in[8];
    const float* W2  = (const float*)d_in[9];
    const float* b2  = (const float*)d_in[10];
    const float* g1  = (const float*)d_in[11];
    const float* be1 = (const float*)d_in[12];
    const float* g2  = (const float*)d_in[13];
    const float* be2 = (const float*)d_in[14];

    const int M = NBATCH * SEQ;  // 16384

    char* ws = (char*)d_ws;
    size_t off = 0;
    auto alloc = [&](size_t bytes) -> void* {
        void* p = ws + off;
        off += (bytes + 255) & ~(size_t)255;
        return p;
    };

    _Float16* h16  = (_Float16*)alloc((size_t)M * DMODEL * 2);
    _Float16* q16  = (_Float16*)alloc((size_t)M * DMODEL * 2);
    _Float16* k16  = (_Float16*)alloc((size_t)M * DMODEL * 2);
    _Float16* vT16 = (_Float16*)alloc((size_t)NBATCH * DMODEL * SEQ * 2);
    float*    sc   = (float*)   alloc((size_t)NBATCH * SEQ * SEQ * 4);
    _Float16* attn = (_Float16*)alloc((size_t)NBATCH * SEQ * SEQ * 2);
    float*    x2   = (float*)   alloc((size_t)M * DMODEL * 4);
    _Float16* h2   = h16;  // h16 dead after QKV GEMMs
    _Float16* h3   = q16;  // q16 dead after scores GEMM

    dim3 blk(256);
    dim3 gMD(DMODEL / BN, M / BM, 1);          // (6,128,1)  [M x 768] GEMMs
    dim3 gSC(SEQ / BN, SEQ / BM, NBATCH);      // (16,16,8)  scores
    dim3 gAV(DMODEL / BN, SEQ / BM, NBATCH);   // (6,16,8)   attn @ V

    // 1) LN1 -> h16 (f16)
    ln_kernel<<<M, blk, 0, stream>>>(x, g1, be1, h16);

    // 2) Q, K, V projections (weights f32 converted in-flight)
    gemm_wmma_kernel<true,true,false,false,1><<<gMD, blk, 0, stream>>>(
        h16, Wq, bq, nullptr, nullptr, q16, M, DMODEL, DMODEL, 0, 0, 0, 0);
    gemm_wmma_kernel<true,true,false,false,1><<<gMD, blk, 0, stream>>>(
        h16, Wk, bk, nullptr, nullptr, k16, M, DMODEL, DMODEL, 0, 0, 0, 0);
    gemm_wmma_kernel<true,true,false,false,2><<<gMD, blk, 0, stream>>>(
        h16, Wv, bv, nullptr, nullptr, vT16, M, DMODEL, DMODEL, 0, 0, 0, 0);

    // 3) scores = q @ k^T (per batch), raw f32 (scale folded into softmax)
    gemm_wmma_kernel<false,false,false,false,0><<<gSC, blk, 0, stream>>>(
        q16, k16, nullptr, nullptr, sc, nullptr, SEQ, SEQ, DMODEL,
        (long)SEQ * DMODEL, (long)SEQ * DMODEL, (long)SEQ * SEQ, 0);

    // 4) softmax rows -> f16 probabilities
    softmax_kernel<<<NBATCH * SEQ, blk, 0, stream>>>(sc, attn);

    // 5) attn @ V + residual x -> x2 (f32)
    gemm_wmma_kernel<false,false,false,true,0><<<gAV, blk, 0, stream>>>(
        attn, vT16, nullptr, x, x2, nullptr, SEQ, DMODEL, SEQ,
        (long)SEQ * SEQ, (long)DMODEL * SEQ, (long)SEQ * DMODEL, (long)SEQ * DMODEL);

    // 6) LN2 -> h2 (f16)
    ln_kernel<<<M, blk, 0, stream>>>(x2, g2, be2, h2);

    // 7) FFN1: gelu(h2 @ W1^T + b1) -> h3 (f16)
    gemm_wmma_kernel<true,true,true,false,1><<<gMD, blk, 0, stream>>>(
        h2, W1, b1, nullptr, nullptr, h3, M, DMODEL, DMODEL, 0, 0, 0, 0);

    // 8) FFN2: gelu(h3 @ W2^T + b2) + x2 -> out (f32)
    gemm_wmma_kernel<true,true,true,true,0><<<gMD, blk, 0, stream>>>(
        h3, W2, b2, x2, (float*)d_out, nullptr, M, DMODEL, DMODEL, 0, 0, 0, 0);
}
